// GCN_Multihead_batch_87385404604421
// MI455X (gfx1250) — matrix operand order repaired and checked by reading, not compiled, not run on hardware
//
#include <hip/hip_runtime.h>
#include <hip/hip_bf16.h>

typedef __attribute__((ext_vector_type(16))) __bf16 v16bf;
typedef __attribute__((ext_vector_type(8)))  __bf16 v8bf;
typedef __attribute__((ext_vector_type(8)))  float  v8f;

#define GB 8
#define GH 4
#define GN 2048
#define GD 128

// ---- kernel 1: column degree d = rsqrt(sum_m adj[b,m,n]) + adj passthrough copy
__global__ __launch_bounds__(256)
void k_degree_copy(const float* __restrict__ adj, float* __restrict__ adj_out,
                   float* __restrict__ deg) {
  const int b = blockIdx.y;
  const int n = blockIdx.x * 256 + threadIdx.x;
  const float* colp = adj + (size_t)b * GN * GN + n;
  float* outp = adj_out + (size_t)b * GN * GN + n;
  float s = 0.f;
  for (int m = 0; m < GN; ++m) {
    float v = colp[(size_t)m * GN];
    outp[(size_t)m * GN] = v;
    s += v;
  }
  deg[b * GN + n] = rsqrtf(s);
}

// ---- tiny passthrough copy (pooled)
__global__ void k_small_copy(const float* __restrict__ src, float* __restrict__ dst, int n) {
  int i = blockIdx.x * 256 + threadIdx.x;
  if (i < n) dst[i] = src[i];
}

// ---- weights -> wT[h][o][i] bf16 (so GEMM1 B-fragments are contiguous in K=i)
__global__ __launch_bounds__(256)
void k_wt(const float* __restrict__ w, __bf16* __restrict__ wT) {
  int idx = blockIdx.x * 256 + threadIdx.x;   // (h*128 + o)*128 + i
  int h = idx >> 14;
  int o = (idx >> 7) & 127;
  int i = idx & 127;
  wT[idx] = (__bf16)w[(h * GD + i) * GD + o];
}

// ---- anorm[b][i][j] = bf16(d_i * a_ij * d_j)
__global__ __launch_bounds__(256)
void k_anorm(const float* __restrict__ adj, const float* __restrict__ deg,
             __bf16* __restrict__ an) {
  size_t e = ((size_t)blockIdx.x * 256 + threadIdx.x) * 8;
  int b = (int)(e >> 22);            // N*N = 2^22
  int i = (int)((e >> 11) & 2047);
  int j = (int)(e & 2047);
  float di = deg[b * GN + i];
  const float4* ap = (const float4*)(adj + e);
  float4 a0 = ap[0], a1 = ap[1];
  const float4* dp = (const float4*)(deg + b * GN + j);
  float4 d0 = dp[0], d1 = dp[1];
  v8bf o;
  o[0] = (__bf16)(di * a0.x * d0.x);
  o[1] = (__bf16)(di * a0.y * d0.y);
  o[2] = (__bf16)(di * a0.z * d0.z);
  o[3] = (__bf16)(di * a0.w * d0.w);
  o[4] = (__bf16)(di * a1.x * d1.x);
  o[5] = (__bf16)(di * a1.y * d1.y);
  o[6] = (__bf16)(di * a1.z * d1.z);
  o[7] = (__bf16)(di * a1.w * d1.w);
  *(v8bf*)(an + e) = o;
}

// ---- GEMM1: supportT[b][h][o][n] = bf16( sum_i x[b,h,n,i] * W[h,i,o] )
// block = 256 thr (8 waves); tile = 128(m=n) x 128(o); K = 128 in steps of 32.
__global__ __launch_bounds__(256)
void k_support(const float* __restrict__ x, const __bf16* __restrict__ wT,
               __bf16* __restrict__ supT) {
  const int b = blockIdx.z, h = blockIdx.y;
  const int m0 = blockIdx.x * 128;
  const int lane = threadIdx.x & 31;
  const int wave = threadIdx.x >> 5;
  const int lh = lane >> 4;      // lane half (K offset group)
  const int l16 = lane & 15;     // row (A) / col (B)

  const float* xrow = x + ((size_t)(b * GH + h) * GN + (m0 + wave * 16 + l16)) * GD;
  const __bf16* wbase = wT + (size_t)h * GD * GD;

  v8f c[8] = {};
#pragma unroll
  for (int kk = 0; kk < GD; kk += 32) {
    // A fragment: row = l16, K = lh*8 + 0..7 and 16 + lh*8 + 0..7 (fp32 -> bf16)
    v16bf a;
    const float* p0 = xrow + kk + lh * 8;
    float4 f0 = *(const float4*)(p0);
    float4 f1 = *(const float4*)(p0 + 4);
    float4 f2 = *(const float4*)(p0 + 16);
    float4 f3 = *(const float4*)(p0 + 20);
    a[0]  = (__bf16)f0.x; a[1]  = (__bf16)f0.y; a[2]  = (__bf16)f0.z; a[3]  = (__bf16)f0.w;
    a[4]  = (__bf16)f1.x; a[5]  = (__bf16)f1.y; a[6]  = (__bf16)f1.z; a[7]  = (__bf16)f1.w;
    a[8]  = (__bf16)f2.x; a[9]  = (__bf16)f2.y; a[10] = (__bf16)f2.z; a[11] = (__bf16)f2.w;
    a[12] = (__bf16)f3.x; a[13] = (__bf16)f3.y; a[14] = (__bf16)f3.z; a[15] = (__bf16)f3.w;
#pragma unroll
    for (int ot = 0; ot < 8; ++ot) {
      // B fragment: col o = ot*16 + l16, K = kk + lh*8 + 0..7 and kk + 16 + lh*8 + 0..7
      const __bf16* bp = wbase + (size_t)(ot * 16 + l16) * GD + kk + lh * 8;
      v8bf blo = *(const v8bf*)bp;
      v8bf bhi = *(const v8bf*)(bp + 16);
      v16bf bfrag = __builtin_shufflevector(blo, bhi, 0, 1, 2, 3, 4, 5, 6, 7,
                                            8, 9, 10, 11, 12, 13, 14, 15);
      c[ot] = __builtin_amdgcn_wmma_f32_16x16x32_bf16(false, a, false, bfrag,
                                                      (short)0, c[ot], false, false);
    }
  }
  // transposed store: supT[(bh*128 + o)*2048 + n]
  const size_t base = (size_t)(b * GH + h) * GD * GN;
  const int nb = m0 + wave * 16 + 8 * lh;
#pragma unroll
  for (int ot = 0; ot < 8; ++ot) {
    const int o = ot * 16 + l16;
#pragma unroll
    for (int r = 0; r < 8; ++r)
      supT[base + (size_t)o * GN + nb + r] = (__bf16)c[ot][r];
  }
}

// ---- GEMM2 + bias + tanh: out[b,h,m,o] = tanh( sum_n anorm[b,m,n]*supportT[b,h,o,n] + bias[h,o] )
// block = 256 thr (8 waves); tile = 128(m) x 128(o); K = 2048 in steps of 32.
// Register-prefetch pipeline: global->regs for slab k+1 issued before the WMMA
// block of slab k, so the global_load latency overlaps 8 WMMAs; single LDS buffer.
__global__ __launch_bounds__(256)
void k_main(const __bf16* __restrict__ an, const __bf16* __restrict__ supT,
            const float* __restrict__ bias, float* __restrict__ out) {
  __shared__ __align__(16) __bf16 lA[128 * 40];   // stride 40 elems (80B): 16B-aligned, conflict-free
  __shared__ __align__(16) __bf16 lB[128 * 40];
  const int b = blockIdx.z, h = blockIdx.y;
  const int m0 = blockIdx.x * 128;
  const int tid = threadIdx.x;
  const int lane = tid & 31, wave = tid >> 5;
  const int lh = lane >> 4, l16 = lane & 15;

  const __bf16* Abase = an + (size_t)b * GN * GN;                  // [m][n]
  const __bf16* Bbase = supT + (size_t)(b * GH + h) * GD * GN;     // [o][n]

  const int sr = tid >> 1;   // staging row 0..127
  const int sh = tid & 1;    // staging 16-elem half of the 32-wide K slab

  // per-thread staging source pointers (advance by 32 bf16 per K-slab)
  const v8bf* gA = (const v8bf*)(Abase + (size_t)(m0 + sr) * GN + sh * 16);
  const v8bf* gB = (const v8bf*)(Bbase + (size_t)sr * GN + sh * 16);
  v8bf* lAst = (v8bf*)(lA + sr * 40 + sh * 16);
  v8bf* lBst = (v8bf*)(lB + sr * 40 + sh * 16);

  float bv[8];
#pragma unroll
  for (int ot = 0; ot < 8; ++ot) bv[ot] = bias[h * GD + ot * 16 + l16];

  // prologue: prefetch slab 0 into registers
  v8bf pa0 = gA[0], pa1 = gA[1];
  v8bf pb0 = gB[0], pb1 = gB[1];

  v8f c[8] = {};
  for (int kk = 0; kk < GN; kk += 32) {
    __syncthreads();                       // previous slab fully consumed
    lAst[0] = pa0; lAst[1] = pa1;          // regs -> LDS
    lBst[0] = pb0; lBst[1] = pb1;
    __syncthreads();                       // tile ready
    if (kk + 32 < GN) {                    // issue next slab's global loads now;
      gA += 4; gB += 4;                    // latency hides under the 8 WMMAs below
      pa0 = gA[0]; pa1 = gA[1];
      pb0 = gB[0]; pb1 = gB[1];
    }
    // A fragment gather (per ISA 16-bit A 16x32 layout)
    v16bf a;
    {
      const __bf16* p = lA + (wave * 16 + l16) * 40 + lh * 8;
      v8bf lo = *(const v8bf*)p;
      v8bf hi = *(const v8bf*)(p + 16);
      a = __builtin_shufflevector(lo, hi, 0, 1, 2, 3, 4, 5, 6, 7,
                                  8, 9, 10, 11, 12, 13, 14, 15);
    }
#pragma unroll
    for (int ot = 0; ot < 8; ++ot) {
      const __bf16* bp = lB + (ot * 16 + l16) * 40 + lh * 8;
      v8bf lo = *(const v8bf*)bp;
      v8bf hi = *(const v8bf*)(bp + 16);
      v16bf bfrag = __builtin_shufflevector(lo, hi, 0, 1, 2, 3, 4, 5, 6, 7,
                                            8, 9, 10, 11, 12, 13, 14, 15);
      c[ot] = __builtin_amdgcn_wmma_f32_16x16x32_bf16(false, a, false, bfrag,
                                                      (short)0, c[ot], false, false);
    }
  }
  // epilogue: bias + tanh, coalesced fp32 stores
  const size_t ob = (size_t)(b * GH + h) * GN * GD;
  const int mr = m0 + wave * 16 + 8 * lh;
#pragma unroll
  for (int ot = 0; ot < 8; ++ot) {
    const int o = ot * 16 + l16;
#pragma unroll
    for (int r = 0; r < 8; ++r)
      out[ob + (size_t)(mr + r) * GD + o] = tanhf(c[ot][r] + bv[ot]);
  }
}

extern "C" void kernel_launch(void* const* d_in, const int* in_sizes, int n_in,
                              void* d_out, int out_size, void* d_ws, size_t ws_size,
                              hipStream_t stream) {
  const float* x      = (const float*)d_in[0];   // [8,4,2048,128]
  const float* adj    = (const float*)d_in[1];   // [8,2048,2048]
  const float* pooled = (const float*)d_in[2];   // [8,128]
  const float* w      = (const float*)d_in[3];   // [4,128,128]
  const float* bias   = (const float*)d_in[4];   // [4,128]

  float* out      = (float*)d_out;                      // [8,4,2048,128]
  float* out_adj  = out + (size_t)GB * GH * GN * GD;    // [8,2048,2048]
  float* out_pool = out_adj + (size_t)GB * GN * GN;     // [8,128]

  // workspace layout (all offsets 256B aligned): ~84 MB total
  char* ws = (char*)d_ws;
  float*  deg  = (float*)(ws);                                              // 64 KB
  __bf16* wT   = (__bf16*)(ws + (64 << 10));                                // 128 KB
  __bf16* supT = (__bf16*)(ws + (64 << 10) + (128 << 10));                  // 16 MB
  __bf16* an   = (__bf16*)(ws + (64 << 10) + (128 << 10) +
                           (size_t)GB * GH * GN * GD * 2);                  // 64 MB

  k_degree_copy<<<dim3(GN / 256, GB), 256, 0, stream>>>(adj, out_adj, deg);
  k_wt<<<(GH * GD * GD) / 256, 256, 0, stream>>>(w, wT);
  k_small_copy<<<(GB * GD + 255) / 256, 256, 0, stream>>>(pooled, out_pool, GB * GD);
  k_anorm<<<(int)((size_t)GB * GN * GN / (8 * 256)), 256, 0, stream>>>(adj, deg, an);
  k_support<<<dim3(GN / 128, GH, GB), 256, 0, stream>>>(x, wT, supT);
  k_main<<<dim3(GN / 128, GH, GB), 256, 0, stream>>>(an, supT, bias, out);
}
